// Model_Q05_Q95_4758823764472
// MI455X (gfx1250) — compile-verified, matrix operand order checked by reading, not run on hardware
//
#include <hip/hip_runtime.h>
#include <cmath>

// ============================================================================
// Informer forward (B=4, L_ENC=2048, L_DEC=1024, D=512, H=8, FF=2048) for
// gfx1250. All GEMM-like work runs through v_wmma_f32_16x16x32_f16:
// QKV/O projections, FFN, QK^T, P@V, distil conv (as 3 accumulating GEMMs),
// and the final 3 projections. ProbSparse attention is computed as dense
// attention (causal for decoder self-attn) to stay on the matrix pipes.
//
// GEMM: 128x64 block tile, 4 waves (32 rows each -> 8 WMMA/k-step), BK=32,
// double-buffered LDS with register-pipelined global loads so HBM latency
// hides behind the WMMAs. All B operands are row-major [N,K] f16 (weights
// transpose-converted once per launch; V written transposed by the permute).
// ============================================================================

typedef _Float16 f16;
typedef __attribute__((ext_vector_type(16))) _Float16 v16h;
typedef __attribute__((ext_vector_type(8)))  _Float16 v8h;
typedef __attribute__((ext_vector_type(4)))  _Float16 v4h;
typedef __attribute__((ext_vector_type(8)))  float    v8f;
typedef __attribute__((ext_vector_type(4)))  float    v4f;

#define NEG_INF (-3.402823466e38f)

__device__ __forceinline__ float geluf(float x) {
  return 0.5f * x * (1.0f + erff(x * 0.70710678118654752f));
}

__device__ __forceinline__ v4h cvt4(v4f a) {
  v4h h; h[0] = (f16)a[0]; h[1] = (f16)a[1]; h[2] = (f16)a[2]; h[3] = (f16)a[3];
  return h;
}

// ---------------------------------------------------------------------------
// Batched WMMA GEMM:  C = act( scale * (A @ B^T_rowmajor) + init )
//   A: f32 row-major [M,K]  (converted to f16 while staging into LDS)
//   B: f16 row-major [N,K]  (i.e. column-major [K,N])
//   initMode: 0 = none, 1 = +bias[col], 2 = accumulate into existing C
//   act: 0 = none, 1 = exact GELU
// Caller guarantees: M multiple of 128; N multiple of 64; K multiple of 32;
// lda multiple of 4; ldb multiple of 8; 16B-aligned base pointers.
// Fragment layout per CDNA5 16x16x32 f16 WMMA: lane (lr=l&15, hf=l>>4) holds
// two contiguous 8-half runs: K = hf*8..+7 and K = 16+hf*8..+7.
// ---------------------------------------------------------------------------
__global__ __launch_bounds__(128) void k_gemm(
    const float* __restrict__ A, const f16* __restrict__ Bm,
    float* __restrict__ C, const float* __restrict__ bias,
    int M, int N, int K, int lda, int ldb, int ldc,
    long long sA, long long sB, long long sC,
    int initMode, int act, float scale)
{
  A  += (long long)blockIdx.z * sA;
  Bm += (long long)blockIdx.z * sB;
  C  += (long long)blockIdx.z * sC;
  const int m0 = blockIdx.y * 128, n0 = blockIdx.x * 64;

  __shared__ alignas(16) f16 As[2][128][40];  // 80B row stride (16B multiple)
  __shared__ alignas(16) f16 Bs[2][64][40];   // Bs[n][k]

  const int tid  = threadIdx.x;
  const int lane = tid & 31, w = tid >> 5;
  const int lr = lane & 15, hf = lane >> 4;

  // Staging coordinates: A = 128 rows x 8 float4-chunks (8 iters/thread),
  //                      B = 64 rows x 4 v8h-chunks (2 iters/thread).
  const int ar = tid >> 3, ac = (tid & 7) * 4;
  const int br = tid >> 2, bc = (tid & 3) * 8;

  // Precomputed per-thread global pointers (stride by k0 only inside loop).
  const float* ap[8];
  #pragma unroll
  for (int i = 0; i < 8; ++i) ap[i] = A + (long long)(m0 + ar + 16 * i) * lda + ac;
  const f16* bp[2];
  #pragma unroll
  for (int i = 0; i < 2; ++i) bp[i] = Bm + (long long)(n0 + br + 32 * i) * ldb + bc;

  v8f acc[2][4] = {};

  // Prologue: stage tile k0=0 into buffer 0.
  #pragma unroll
  for (int i = 0; i < 8; ++i) *(v4h*)&As[0][ar + 16 * i][ac] = cvt4(*(const v4f*)ap[i]);
  #pragma unroll
  for (int i = 0; i < 2; ++i) *(v8h*)&Bs[0][br + 32 * i][bc] = *(const v8h*)bp[i];

  int p = 0;
  for (int k0 = 0; k0 < K; k0 += 32) {
    __syncthreads();
    const bool more = (k0 + 32) < K;

    // Issue next tile's global loads early; latency hides behind the WMMAs.
    v4f an[8]; v8h bn[2];
    if (more) {
      #pragma unroll
      for (int i = 0; i < 8; ++i) an[i] = *(const v4f*)(ap[i] + k0 + 32);
      #pragma unroll
      for (int i = 0; i < 2; ++i) bn[i] = *(const v8h*)(bp[i] + k0 + 32);
    }

    // Compute from buffer p: 2 A-frags x 4 B-frags = 8 WMMAs.
    #pragma unroll
    for (int mt = 0; mt < 2; ++mt) {
      const int arow = w * 32 + mt * 16 + lr;
      v8h a0 = *(const v8h*)&As[p][arow][hf * 8];
      v8h a1 = *(const v8h*)&As[p][arow][16 + hf * 8];
      v16h av = __builtin_shufflevector(a0, a1, 0,1,2,3,4,5,6,7,8,9,10,11,12,13,14,15);
      #pragma unroll
      for (int nt = 0; nt < 4; ++nt) {
        v8h b0 = *(const v8h*)&Bs[p][nt * 16 + lr][hf * 8];
        v8h b1 = *(const v8h*)&Bs[p][nt * 16 + lr][16 + hf * 8];
        v16h bv = __builtin_shufflevector(b0, b1, 0,1,2,3,4,5,6,7,8,9,10,11,12,13,14,15);
        acc[mt][nt] = __builtin_amdgcn_wmma_f32_16x16x32_f16(
            false, av, false, bv, (short)0, acc[mt][nt], false, false);
      }
    }

    // Store the staged registers into the other buffer.
    if (more) {
      #pragma unroll
      for (int i = 0; i < 8; ++i) *(v4h*)&As[1 - p][ar + 16 * i][ac] = cvt4(an[i]);
      #pragma unroll
      for (int i = 0; i < 2; ++i) *(v8h*)&Bs[1 - p][br + 32 * i][bc] = bn[i];
    }
    p ^= 1;
  }

  // Epilogue. C/D frag: lane l -> col lr, VGPR r -> row r + 8*hf.
  #pragma unroll
  for (int mt = 0; mt < 2; ++mt) {
    #pragma unroll
    for (int nt = 0; nt < 4; ++nt) {
      const int col = n0 + nt * 16 + lr;
      const float bv = (initMode == 1 && bias) ? bias[col] : 0.0f;
      #pragma unroll
      for (int r = 0; r < 8; ++r) {
        const int row = m0 + w * 32 + mt * 16 + r + 8 * hf;
        const long long off = (long long)row * ldc + col;
        float v = scale * acc[mt][nt][r];
        v += (initMode == 2) ? C[off] : bv;
        if (act == 1) v = geluf(v);
        C[off] = v;
      }
    }
  }
}

// ---------------------------------------------------------------------------
// Transpose-convert: src f32 [K,N] row-major -> dst f16 [N,K] row-major.
__global__ void k_cvt_t(const float* __restrict__ s, f16* __restrict__ d,
                        int K, int N) {
  long long i = (long long)blockIdx.x * 256 + threadIdx.x;  // output index [n][k]
  if (i >= (long long)K * N) return;
  int k = (int)(i % K), n = (int)(i / K);
  d[i] = (f16)s[(long long)k * N + n];
}

// [B,L,H,Dh] f32 -> [B,H,L,Dh] (T = float for Q, f16 for K)
template <typename T>
__global__ void k_permute(const float* __restrict__ in, T* __restrict__ out,
                          int Bn, int L, int H, int Dh) {
  long long i = (long long)blockIdx.x * 256 + threadIdx.x;
  long long total = (long long)Bn * L * H * Dh;
  if (i >= total) return;
  int d = (int)(i % Dh); long long t = i / Dh;
  int h = (int)(t % H);  t /= H;
  int l = (int)(t % L);  int b = (int)(t / L);
  out[(((long long)b * H + h) * L + l) * Dh + d] = (T)in[i];
}

// [B,L,H,Dh] f32 -> V^T f16 [B,H,Dh,L]  (so P@V uses the [N,K] B path)
__global__ void k_permute_vt(const float* __restrict__ in, f16* __restrict__ out,
                             int Bn, int L, int H, int Dh) {
  long long i = (long long)blockIdx.x * 256 + threadIdx.x;
  long long total = (long long)Bn * L * H * Dh;
  if (i >= total) return;
  int d = (int)(i % Dh); long long t = i / Dh;
  int h = (int)(t % H);  t /= H;
  int l = (int)(t % L);  int b = (int)(t / L);
  out[(((long long)b * H + h) * Dh + d) * L + l] = (f16)in[i];
}

// ctx [B,H,L,Dh] -> out [B,L,H*Dh]; mix=1 reinterprets flat (Informer quirk)
__global__ void k_merge(const float* __restrict__ ctx, float* __restrict__ out,
                        int Bn, int L, int H, int Dh, int mix) {
  long long i = (long long)blockIdx.x * 256 + threadIdx.x;
  long long total = (long long)Bn * L * H * Dh;
  if (i >= total) return;
  if (mix) { out[i] = ctx[i]; return; }
  int d = (int)(i % Dh); long long t = i / Dh;
  int h = (int)(t % H);  t /= H;
  int l = (int)(t % L);  int b = (int)(t / L);
  out[i] = ctx[(((long long)b * H + h) * L + l) * Dh + d];
}

// Row softmax over S[row, 0..cols) with optional causal mask (col > qglob).
__global__ __launch_bounds__(256) void k_softmax(float* __restrict__ S,
                                                 int cols, int mc, int m0, int causal) {
  long long row = blockIdx.x;
  float* p = S + row * (long long)cols;
  const int qglob = m0 + (int)(row % mc);
  __shared__ float red[256];
  float mx = NEG_INF;
  for (int j = threadIdx.x; j < cols; j += 256) {
    float v = p[j];
    if (causal && j > qglob) v = NEG_INF;
    mx = fmaxf(mx, v);
  }
  red[threadIdx.x] = mx; __syncthreads();
  for (int s = 128; s > 0; s >>= 1) {
    if (threadIdx.x < (unsigned)s) red[threadIdx.x] = fmaxf(red[threadIdx.x], red[threadIdx.x + s]);
    __syncthreads();
  }
  mx = red[0]; __syncthreads();
  float sum = 0.0f;
  for (int j = threadIdx.x; j < cols; j += 256) {
    float v = p[j];
    v = (causal && j > qglob) ? 0.0f : expf(v - mx);
    p[j] = v; sum += v;
  }
  red[threadIdx.x] = sum; __syncthreads();
  for (int s = 128; s > 0; s >>= 1) {
    if (threadIdx.x < (unsigned)s) red[threadIdx.x] += red[threadIdx.x + s];
    __syncthreads();
  }
  const float inv = 1.0f / red[0];
  for (int j = threadIdx.x; j < cols; j += 256) p[j] *= inv;
}

// out = LayerNorm(X + Y) ; Y==nullptr -> plain LayerNorm(X). One block per row.
__global__ __launch_bounds__(256) void k_add_ln(
    const float* __restrict__ X, const float* __restrict__ Y,
    const float* __restrict__ g, const float* __restrict__ b,
    float* __restrict__ out, int D) {
  long long row = blockIdx.x;
  const float* x = X + row * (long long)D;
  const float* y = Y ? Y + row * (long long)D : nullptr;
  float* o = out + row * (long long)D;
  __shared__ float rs[256], rq[256];
  float s = 0.0f, s2 = 0.0f;
  for (int j = threadIdx.x; j < D; j += 256) {
    float v = x[j] + (y ? y[j] : 0.0f);
    s += v; s2 += v * v;
  }
  rs[threadIdx.x] = s; rq[threadIdx.x] = s2; __syncthreads();
  for (int t = 128; t > 0; t >>= 1) {
    if (threadIdx.x < (unsigned)t) { rs[threadIdx.x] += rs[threadIdx.x + t]; rq[threadIdx.x] += rq[threadIdx.x + t]; }
    __syncthreads();
  }
  const float mean = rs[0] / (float)D;
  const float var  = rq[0] / (float)D - mean * mean;
  const float inv  = rsqrtf(var + 1e-5f);
  for (int j = threadIdx.x; j < D; j += 256) {
    float v = x[j] + (y ? y[j] : 0.0f);
    o[j] = g[j] * (v - mean) * inv + b[j];
  }
}

// Fused data embedding: circular conv (K=3, Cin small) + sinusoidal PE +
// x_mark @ time_w + time_b.
__global__ void k_embed(const float* __restrict__ x, const float* __restrict__ mark,
                        const float* __restrict__ cw, const float* __restrict__ cb,
                        const float* __restrict__ tw, const float* __restrict__ tb,
                        float* __restrict__ out, int Bn, int L, int Cin, int D) {
  long long i = (long long)blockIdx.x * 256 + threadIdx.x;
  long long total = (long long)Bn * L * D;
  if (i >= total) return;
  int d = (int)(i % D); long long t = i / D;
  int l = (int)(t % L); int b = (int)(t / L);
  float acc = cb[d];
  #pragma unroll
  for (int tt = 0; tt < 3; ++tt) {
    int ls = l - 1 + tt;
    ls = (ls < 0) ? L - 1 : ((ls >= L) ? 0 : ls);
    const float* xr = x + ((long long)b * L + ls) * Cin;
    for (int ci = 0; ci < Cin; ++ci)
      acc += xr[ci] * cw[((long long)(tt * Cin + ci)) * D + d];
  }
  const int i2 = d >> 1;
  const float div = expf(-(float)(2 * i2) * (logf(10000.0f) / (float)D));
  const float ang = (float)l * div;
  acc += (d & 1) ? cosf(ang) : sinf(ang);
  const float* mr = mark + ((long long)b * L + l) * 4;
  #pragma unroll
  for (int m = 0; m < 4; ++m) acc += mr[m] * tw[m * D + d];
  acc += tb[d];
  out[i] = acc;
}

// Circular pad along L: xp[B, L+2, D]
__global__ void k_circpad(const float* __restrict__ x, float* __restrict__ xp,
                          int Bn, int L, int D) {
  long long i = (long long)blockIdx.x * 256 + threadIdx.x;
  long long total = (long long)Bn * (L + 2) * D;
  if (i >= total) return;
  int d = (int)(i % D); long long t = i / D;
  int l = (int)(t % (L + 2)); int b = (int)(t / (L + 2));
  int ls = (l == 0) ? L - 1 : ((l == L + 1) ? 0 : l - 1);
  xp[i] = x[((long long)b * L + ls) * D + d];
}

// h = elu( bn_g * h / sqrt(1+eps) + bn_b )
__global__ void k_bn_elu(float* __restrict__ h, const float* __restrict__ g,
                         const float* __restrict__ b, long long n, int D) {
  long long i = (long long)blockIdx.x * 256 + threadIdx.x;
  if (i >= n) return;
  int d = (int)(i % D);
  float v = h[i] * (1.0f / sqrtf(1.0f + 1e-5f)) * g[d] + b[d];
  h[i] = (v > 0.0f) ? v : expm1f(v);
}

// maxpool over L: window 3, stride 2, pad (1,1) -> Lout = L/2
__global__ void k_maxpool(const float* __restrict__ h, float* __restrict__ out,
                          int Bn, int L, int D) {
  const int Lo = L / 2;
  long long i = (long long)blockIdx.x * 256 + threadIdx.x;
  long long total = (long long)Bn * Lo * D;
  if (i >= total) return;
  int d = (int)(i % D); long long t = i / D;
  int lo = (int)(t % Lo); int b = (int)(t / Lo);
  float m = NEG_INF;
  #pragma unroll
  for (int dl = -1; dl <= 1; ++dl) {
    int li = 2 * lo + dl;
    if (li >= 0 && li < L) m = fmaxf(m, h[((long long)b * L + li) * D + d]);
  }
  out[i] = m;
}

// ============================================================================
// Host orchestration
// ============================================================================
extern "C" void kernel_launch(void* const* d_in, const int* in_sizes, int n_in,
                              void* d_out, int out_size, void* d_ws, size_t ws_size,
                              hipStream_t stream)
{
  (void)in_sizes; (void)out_size; (void)ws_size;
  if (n_in < 84) return;
  auto F = [&](int i) { return (const float*)d_in[i]; };

  // --- deterministic bump allocator over d_ws ---
  size_t off = 0; char* base = (char*)d_ws;
  auto alloc = [&](size_t bytes) -> void* {
    size_t a = (off + 255) & ~(size_t)255; off = a + bytes; return (void*)(base + a);
  };
  auto af = [&](size_t n) { return (float*)alloc(n * sizeof(float)); };
  auto ah = [&](size_t n) { return (f16*)alloc(n * sizeof(f16)); };

  // Transpose-convert a weight [K,N] f32 -> [N,K] f16.
  auto cvtT = [&](const float* w, int K, int N) -> f16* {
    f16* d = ah((size_t)K * N);
    long long n = (long long)K * N;
    k_cvt_t<<<dim3((unsigned)((n + 255) / 256)), dim3(256), 0, stream>>>(w, d, K, N);
    return d;
  };

  auto gemm = [&](const float* A, const f16* Bw, float* C, const float* bias,
                  int M, int N, int K, int lda, int ldb, int ldc,
                  long long sA, long long sB, long long sC, int batch,
                  int initMode, int act, float scale) {
    dim3 g((unsigned)(N / 64), (unsigned)(M / 128), (unsigned)batch);
    k_gemm<<<g, dim3(128), 0, stream>>>(A, Bw, C, bias, M, N, K, lda, ldb, ldc,
                                        sA, sB, sC, initMode, act, scale);
  };

  const int Bn = 4, Le = 2048, Lm = 1024, D = 512, H = 8, Dh = 64, FF = 2048, MC = 256;

  // --- input index map (depth-first over setup_inputs dict order) ---
  // 0..3: x_enc, x_mark_enc, x_dec, x_mark_dec
  // 4..7: enc_emb{conv_w,conv_b,time_w,time_b}; 8..11: dec_emb{...}
  // enc layer i base=12+16i: w1,b1,w2,b2,n1g,n1b,n2g,n2b, wq,bq,wk,bk,wv,bv,wo,bo
  // 44..47: enc_conv{w,b,bn_g,bn_b}; 48,49: enc_norm{g,b}
  // 50..75: dec layer (ffn 8, self_attn 8, cross_attn 8, n3g,n3b)
  // 76,77: dec_norm; 78,79: proj; 80,81: proj_q05; 82,83: proj_q95

  struct AttnW { f16 *wq, *wk, *wv, *wo; const float *bq, *bk, *bv, *bo; };
  auto cvt_attn = [&](int bidx) {
    AttnW a;
    a.wq = cvtT(F(bidx + 0), D, D); a.bq = F(bidx + 1);
    a.wk = cvtT(F(bidx + 2), D, D); a.bk = F(bidx + 3);
    a.wv = cvtT(F(bidx + 4), D, D); a.bv = F(bidx + 5);
    a.wo = cvtT(F(bidx + 6), D, D); a.bo = F(bidx + 7);
    return a;
  };

  // Weight conversions (all before activations; same stream => ordered).
  f16 *enc_w1[2], *enc_w2[2]; AttnW enc_at[2];
  for (int i = 0; i < 2; ++i) {
    int bidx = 12 + 16 * i;
    enc_w1[i] = cvtT(F(bidx + 0), D, FF);     // -> [FF, D], ldb = D
    enc_w2[i] = cvtT(F(bidx + 2), FF, D);     // -> [D, FF], ldb = FF
    enc_at[i] = cvt_attn(bidx + 8);
  }
  f16* cw16[3];
  for (int t = 0; t < 3; ++t) cw16[t] = cvtT(F(44) + (size_t)t * D * D, D, D);
  f16* dec_w1 = cvtT(F(50), D, FF);
  f16* dec_w2 = cvtT(F(52), FF, D);
  AttnW dec_sa = cvt_attn(58);
  AttnW dec_ca = cvt_attn(66);
  f16* proj_w = cvtT(F(78), D, D);
  f16* q05_w  = cvtT(F(80), D, D);
  f16* q95_w  = cvtT(F(82), D, D);

  // Activation buffers.
  float* encx  = af((size_t)Bn * Le * D);
  float* attno = af((size_t)Bn * Le * D);
  float* xtmp  = af((size_t)Bn * Le * D);
  float* hbuf  = af((size_t)Bn * Le * FF);
  float* ybuf  = af((size_t)Bn * Le * D);
  float* qlin  = af((size_t)Bn * Le * D);
  float* klin  = af((size_t)Bn * Le * D);
  float* vlin  = af((size_t)Bn * Le * D);
  float* Qp    = af((size_t)Bn * Le * D);
  float* ctx   = af((size_t)Bn * Le * D);
  float* mrg   = af((size_t)Bn * Le * D);
  float* Sbuf  = af((size_t)Bn * H * MC * Le);
  f16*   K16   = ah((size_t)Bn * Le * D);
  f16*   V16T  = ah((size_t)Bn * Le * D);
  float* xp    = af((size_t)Bn * (Le + 2) * D);
  float* cv    = af((size_t)Bn * Le * D);
  float* encx2 = af((size_t)Bn * Lm * D);
  float* encout= af((size_t)Bn * Lm * D);
  float* decx  = af((size_t)Bn * Lm * D);
  float* dbuf1 = af((size_t)Bn * Lm * D);
  float* dbuf2 = af((size_t)Bn * Lm * D);
  float* dnorm = af((size_t)Bn * Lm * D);
  float* outp  = af((size_t)Bn * 512 * D);

  auto grid1 = [&](long long n) { return dim3((unsigned)((n + 255) / 256)); };

  // Dense attention block: out = (merge(softmax(Q K^T / sqrt(Dh)) V)) Wo + bo
  auto attention = [&](const float* xq, const float* xkv, int Lq, int Lkv,
                       const AttnW& aw, int causal, int mix, float* out) {
    gemm(xq,  aw.wq, qlin, aw.bq, Bn * Lq,  D, D, D, D, D, 0, 0, 0, 1, 1, 0, 1.0f);
    gemm(xkv, aw.wk, klin, aw.bk, Bn * Lkv, D, D, D, D, D, 0, 0, 0, 1, 1, 0, 1.0f);
    gemm(xkv, aw.wv, vlin, aw.bv, Bn * Lkv, D, D, D, D, D, 0, 0, 0, 1, 1, 0, 1.0f);
    long long tq = (long long)Bn * Lq * D, tk = (long long)Bn * Lkv * D;
    k_permute<float><<<grid1(tq), dim3(256), 0, stream>>>(qlin, Qp,  Bn, Lq,  H, Dh);
    k_permute<f16>  <<<grid1(tk), dim3(256), 0, stream>>>(klin, K16, Bn, Lkv, H, Dh);
    k_permute_vt    <<<grid1(tk), dim3(256), 0, stream>>>(vlin, V16T, Bn, Lkv, H, Dh);
    for (int m0 = 0; m0 < Lq; m0 += MC) {
      // S = Q[m0:m0+MC] @ K^T * 1/sqrt(Dh)   (batch = B*H; K16 is [Lkv,Dh])
      gemm(Qp + (long long)m0 * Dh, K16, Sbuf, nullptr,
           MC, Lkv, Dh, Dh, Dh, Lkv,
           (long long)Lq * Dh, (long long)Lkv * Dh, (long long)MC * Lkv, Bn * H,
           0, 0, 0.125f);
      k_softmax<<<dim3((unsigned)(Bn * H * MC)), dim3(256), 0, stream>>>(Sbuf, Lkv, MC, m0, causal);
      // ctx[m0:m0+MC] = P @ V    (V16T is [Dh,Lkv] => B row-major [N,K])
      gemm(Sbuf, V16T, ctx + (long long)m0 * Dh, nullptr,
           MC, Dh, Lkv, Lkv, Lkv, Dh,
           (long long)MC * Lkv, (long long)Dh * Lkv, (long long)Lq * Dh, Bn * H,
           0, 0, 1.0f);
    }
    k_merge<<<grid1(tq), dim3(256), 0, stream>>>(ctx, mrg, Bn, Lq, H, Dh, mix);
    gemm(mrg, aw.wo, out, aw.bo, Bn * Lq, D, D, D, D, D, 0, 0, 0, 1, 1, 0, 1.0f);
  };

  auto enc_layer = [&](float* x, int L, int li) {
    const int bidx = 12 + 16 * li;
    attention(x, x, L, L, enc_at[li], 0, 0, attno);
    k_add_ln<<<dim3((unsigned)(Bn * L)), dim3(256), 0, stream>>>(x, attno, F(bidx + 4), F(bidx + 5), xtmp, D);
    gemm(xtmp, enc_w1[li], hbuf, F(bidx + 1), Bn * L, FF, D, D, D, FF, 0, 0, 0, 1, 1, 1, 1.0f);
    gemm(hbuf, enc_w2[li], ybuf, F(bidx + 3), Bn * L, D, FF, FF, FF, D, 0, 0, 0, 1, 1, 0, 1.0f);
    k_add_ln<<<dim3((unsigned)(Bn * L)), dim3(256), 0, stream>>>(xtmp, ybuf, F(bidx + 6), F(bidx + 7), x, D);
  };

  // ------------------- encoder -------------------
  k_embed<<<grid1((long long)Bn * Le * D), dim3(256), 0, stream>>>(
      F(0), F(1), F(4), F(5), F(6), F(7), encx, Bn, Le, 7, D);
  enc_layer(encx, Le, 0);

  // distil: circular-pad + conv-as-3-GEMMs + BN/ELU + maxpool(3,2,pad1)
  k_circpad<<<grid1((long long)Bn * (Le + 2) * D), dim3(256), 0, stream>>>(encx, xp, Bn, Le, D);
  for (int t = 0; t < 3; ++t)
    gemm(xp + (long long)t * D, cw16[t], cv, (t == 0) ? F(45) : nullptr,
         Le, D, D, D, D, D,
         (long long)(Le + 2) * D, 0, (long long)Le * D, Bn,
         (t == 0) ? 1 : 2, 0, 1.0f);
  k_bn_elu<<<grid1((long long)Bn * Le * D), dim3(256), 0, stream>>>(cv, F(46), F(47), (long long)Bn * Le * D, D);
  k_maxpool<<<grid1((long long)Bn * Lm * D), dim3(256), 0, stream>>>(cv, encx2, Bn, Le, D);

  enc_layer(encx2, Lm, 1);
  k_add_ln<<<dim3((unsigned)(Bn * Lm)), dim3(256), 0, stream>>>(encx2, nullptr, F(48), F(49), encout, D);

  // ------------------- decoder -------------------
  k_embed<<<grid1((long long)Bn * Lm * D), dim3(256), 0, stream>>>(
      F(2), F(3), F(8), F(9), F(10), F(11), decx, Bn, Lm, 7, D);

  attention(decx, decx, Lm, Lm, dec_sa, /*causal*/1, /*mix*/1, attno);
  k_add_ln<<<dim3((unsigned)(Bn * Lm)), dim3(256), 0, stream>>>(decx, attno, F(54), F(55), dbuf1, D);
  attention(dbuf1, encout, Lm, Lm, dec_ca, 0, 0, attno);
  k_add_ln<<<dim3((unsigned)(Bn * Lm)), dim3(256), 0, stream>>>(dbuf1, attno, F(56), F(57), dbuf2, D);
  gemm(dbuf2, dec_w1, hbuf, F(51), Bn * Lm, FF, D, D, D, FF, 0, 0, 0, 1, 1, 1, 1.0f);
  gemm(hbuf, dec_w2, ybuf, F(53), Bn * Lm, D, FF, FF, FF, D, 0, 0, 0, 1, 1, 0, 1.0f);
  k_add_ln<<<dim3((unsigned)(Bn * Lm)), dim3(256), 0, stream>>>(dbuf2, ybuf, F(74), F(75), decx, D);
  k_add_ln<<<dim3((unsigned)(Bn * Lm)), dim3(256), 0, stream>>>(decx, nullptr, F(76), F(77), dnorm, D);

  // ------------------- projections (only last PRED_LEN=512 rows matter) ----
  gemm(dnorm + (long long)512 * D, proj_w, outp, F(79),
       512, D, D, D, D, D,
       (long long)Lm * D, 0, (long long)512 * D, Bn,
       1, 0, 1.0f);
  float* OUT = (float*)d_out;
  gemm(outp, q05_w, OUT, F(81), Bn * 512, D, D, D, D, D, 0, 0, 0, 1, 1, 0, 1.0f);
  gemm(outp, q95_w, OUT + (long long)Bn * 512 * D, F(83), Bn * 512, D, D, D, D, D, 0, 0, 0, 1, 1, 0, 1.0f);
}
